// MiniGPT_40922448396374
// MI455X (gfx1250) — compile-verified, matrix operand order checked by reading, not run on hardware
//
#include <hip/hip_runtime.h>
#include <hip/hip_bf16.h>

typedef __attribute__((ext_vector_type(16))) __bf16 v16bf;
typedef __attribute__((ext_vector_type(8)))  float  v8f;

#define Bc 8
#define Tc 1024
#define Vc 8192
#define Dc 768
#define Hc 12
#define Lc 6
#define DHc 64
#define FFc 3072
#define Mrows (Bc * Tc)          // 8192

#define GEMM_SMEM_BYTES (64 * 40 * 2 + 32 * 136 * 2)   // sA 5120B + sB 8704B

__device__ __forceinline__ unsigned short f2bf(float f) {
    unsigned u = __builtin_bit_cast(unsigned, f);
    unsigned r = u + 0x7FFFu + ((u >> 16) & 1u);
    return (unsigned short)(r >> 16);
}

__device__ __forceinline__ void wait_async0() {
#if __has_builtin(__builtin_amdgcn_s_wait_asynccnt)
    __builtin_amdgcn_s_wait_asynccnt(0);
#else
    asm volatile("s_wait_asynccnt 0x0" ::: "memory");
#endif
}

// Async LDS-DMA: copy 16 bytes from global -> LDS (ASYNCcnt path, CDNA5).
// ldsOff is a byte offset from the wave's LDS base (dynamic LDS starts at 0).
__device__ __forceinline__ void async_copy_b128(unsigned ldsOff, const void* gsrc) {
    unsigned long long ga = (unsigned long long)gsrc;
    asm volatile("global_load_async_to_lds_b128 %0, %1, off"
                 :: "v"(ldsOff), "v"(ga)
                 : "memory");
}

// Load one 16-element bf16 WMMA operand fragment for this lane.
// p points at this lane's k=0 element; k-th element is p[k*kstride].
// kgrp = (lane>=16 ? 8 : 0); K interleave per CDNA5 ISA 7.12.2 (16-bit A/B 16x32).
__device__ __forceinline__ v16bf load_frag(const unsigned short* p, int kstride, int kgrp) {
    v16bf f;
#pragma unroll
    for (int j = 0; j < 16; ++j) {
        int kk = kgrp + (j < 8 ? j : j + 8);
        f[j] = __builtin_bit_cast(__bf16, p[kk * kstride]);
    }
    return f;
}

// ---------------------------------------------------------------------------
// Embedding: x[b,t,:] = tok_emb[idx[b,t],:] + pos_emb[t,:]
// ---------------------------------------------------------------------------
__global__ __launch_bounds__(256) void embed_kernel(const int* __restrict__ idx,
                                                    const float* __restrict__ tok,
                                                    const float* __restrict__ pos,
                                                    float* __restrict__ x) {
    size_t i = (size_t)blockIdx.x * 256 + threadIdx.x;
    if (i >= (size_t)Mrows * Dc) return;
    int d = (int)(i % Dc);
    size_t bt = i / Dc;
    int t = (int)(bt % Tc);
    x[i] = tok[(size_t)idx[bt] * Dc + d] + pos[(size_t)t * Dc + d];
}

// ---------------------------------------------------------------------------
// LayerNorm over D=768, writes bf16. One block per row.
// ---------------------------------------------------------------------------
__global__ __launch_bounds__(256) void ln_kernel(const float* __restrict__ x,
                                                 const float* __restrict__ g,
                                                 const float* __restrict__ beta,
                                                 unsigned short* __restrict__ out) {
    __shared__ float red[256];
    int tid = threadIdx.x;
    size_t row = blockIdx.x;
    const float* xr = x + row * Dc;

    float s = 0.f;
    for (int i = tid; i < Dc; i += 256) s += xr[i];
    red[tid] = s; __syncthreads();
    for (int o = 128; o > 0; o >>= 1) { if (tid < o) red[tid] += red[tid + o]; __syncthreads(); }
    float mean = red[0] * (1.0f / Dc);
    __syncthreads();

    float sv = 0.f;
    for (int i = tid; i < Dc; i += 256) { float d = xr[i] - mean; sv += d * d; }
    red[tid] = sv; __syncthreads();
    for (int o = 128; o > 0; o >>= 1) { if (tid < o) red[tid] += red[tid + o]; __syncthreads(); }
    float inv = rsqrtf(red[0] * (1.0f / Dc) + 1e-5f);

    for (int i = tid; i < Dc; i += 256)
        out[row * Dc + i] = f2bf((xr[i] - mean) * inv * g[i] + beta[i]);
}

// ---------------------------------------------------------------------------
// bf16 WMMA GEMM: out[M,N] = A[M,K](bf16) * W[K,N](f32->bf16) (+ bias)
// WMODE 0: W is [K,N] row-major. WMODE 1: QKV weights [H, K, 64], n = h*64+e.
// EMODE 0: f32 store. 1: f32 residual in-place add. 2: relu->bf16. 3: bf16.
// Block: 256 thr (8 waves), tile 64x128; wave tile 32x32 (4 WMMAs / K-step).
// A tile staged into LDS via GLOBAL_LOAD_ASYNC_TO_LDS_B128 (ASYNCcnt DMA),
// overlapping the VALU f32->bf16 conversion that stages the B tile.
// Uses dynamic LDS only, so LDS byte offsets start at 0 for the async VDST.
// ---------------------------------------------------------------------------
template <int WMODE, int EMODE>
__global__ __launch_bounds__(256) void gemm_wmma(const unsigned short* __restrict__ A,
                                                 const float* __restrict__ W,
                                                 const float* __restrict__ bias,
                                                 float* __restrict__ outF,
                                                 unsigned short* __restrict__ outB,
                                                 int M, int N, int K) {
    extern __shared__ char smem[];
    unsigned short (*sA)[40]  = reinterpret_cast<unsigned short(*)[40]>(smem);          // 64x32+pad
    unsigned short (*sB)[136] = reinterpret_cast<unsigned short(*)[136]>(smem + 5120);  // 32x128+pad

    int tid  = threadIdx.x;
    int m0   = blockIdx.y * 64;
    int n0   = blockIdx.x * 128;
    int lane = tid & 31;
    int wid  = tid >> 5;
    int wm   = wid >> 2;        // 0..1
    int wn   = wid & 3;         // 0..3
    int lrow = lane & 15;
    int kgrp = (lane >> 4) << 3;

    // Per-thread async-copy assignment: one b128 (8 bf16) of the 64x32 A tile.
    int ar  = tid >> 2;          // 0..63  : tile row
    int ac8 = (tid & 3) * 8;     // 0/8/16/24 : col of 8-element chunk
    unsigned ldsAoff = (unsigned)((ar * 40 + ac8) * 2);   // bytes, 16B aligned

    v8f z = {};
    v8f acc[2][2];
    acc[0][0] = z; acc[0][1] = z; acc[1][0] = z; acc[1][1] = z;

    for (int k0 = 0; k0 < K; k0 += 32) {
        // Kick off async LDS-DMA for the A tile (bf16 bytes, no conversion).
        async_copy_b128(ldsAoff, &A[(size_t)(m0 + ar) * K + (k0 + ac8)]);

        // Meanwhile stage B tile through VGPRs with f32 -> bf16 convert.
        for (int i = tid; i < 32 * 128; i += 256) {
            int r = i >> 7, c = i & 127;
            int kk = k0 + r, nn = n0 + c;
            float w = (WMODE == 0)
                          ? W[(size_t)kk * N + nn]
                          : W[((size_t)(nn >> 6) * K + kk) * 64 + (nn & 63)];
            sB[r][c] = f2bf(w);
        }
        if (k0 + 32 < K) {  // prefetch next K tile (global_prefetch_b8)
            __builtin_prefetch(&A[(size_t)(m0 + (tid & 63)) * K + (k0 + 32)]);
            if (WMODE == 0)
                __builtin_prefetch(&W[(size_t)(k0 + 32 + (tid & 31)) * N + n0]);
        }

        wait_async0();       // this wave's async writes to LDS have landed
        __syncthreads();     // publish A (async) + B (ds_store) to all waves

        v16bf a0 = load_frag(&sA[wm * 32 + lrow][0],      1, kgrp);
        v16bf a1 = load_frag(&sA[wm * 32 + 16 + lrow][0], 1, kgrp);
        v16bf b0 = load_frag(&sB[0][wn * 32 + lrow],      136, kgrp);
        v16bf b1 = load_frag(&sB[0][wn * 32 + 16 + lrow], 136, kgrp);

        acc[0][0] = __builtin_amdgcn_wmma_f32_16x16x32_bf16(false, a0, false, b0, (short)0, acc[0][0], false, false);
        acc[0][1] = __builtin_amdgcn_wmma_f32_16x16x32_bf16(false, a0, false, b1, (short)0, acc[0][1], false, false);
        acc[1][0] = __builtin_amdgcn_wmma_f32_16x16x32_bf16(false, a1, false, b0, (short)0, acc[1][0], false, false);
        acc[1][1] = __builtin_amdgcn_wmma_f32_16x16x32_bf16(false, a1, false, b1, (short)0, acc[1][1], false, false);
        __syncthreads();     // all waves done reading before next tile lands
    }

    // Epilogue: C/D layout = 8 rows (r + 8*(lane>=16)) x col (lane&15)
#pragma unroll
    for (int mi = 0; mi < 2; ++mi)
#pragma unroll
        for (int ni = 0; ni < 2; ++ni) {
            int ncol = n0 + wn * 32 + ni * 16 + lrow;
            float bval = bias ? bias[ncol] : 0.f;
#pragma unroll
            for (int r = 0; r < 8; ++r) {
                int mrow = m0 + wm * 32 + mi * 16 + r + kgrp;
                float val = acc[mi][ni][r] + bval;
                size_t oi = (size_t)mrow * N + ncol;
                if (EMODE == 0)      outF[oi] = val;
                else if (EMODE == 1) outF[oi] = outF[oi] + val;
                else if (EMODE == 2) outB[oi] = f2bf(fmaxf(val, 0.f));
                else                 outB[oi] = f2bf(val);
            }
        }
}

// ---------------------------------------------------------------------------
// Flash attention: one wave per (b,h,16-query tile). Online softmax.
// q/k/v/o are bf16 [B,T,D] (heads concatenated); scale = D^-0.5 per reference.
// ---------------------------------------------------------------------------
__global__ __launch_bounds__(32) void attn_kernel(const unsigned short* __restrict__ q,
                                                  const unsigned short* __restrict__ k,
                                                  const unsigned short* __restrict__ v,
                                                  unsigned short* __restrict__ o) {
    __shared__ unsigned short sP[16][32];
    int lane = threadIdx.x & 31;
    int bh = blockIdx.y;
    int b = bh / Hc, h = bh % Hc;
    int qr = blockIdx.x * 16;
    int lrow = lane & 15;
    int kgrp = (lane >> 4) << 3;   // 0 or 8; also the row offset of this half

    const unsigned short* qp = q + (size_t)b * Tc * Dc + h * DHc;
    const unsigned short* kp = k + (size_t)b * Tc * Dc + h * DHc;
    const unsigned short* vp = v + (size_t)b * Tc * Dc + h * DHc;

    v16bf aq0 = load_frag(qp + (size_t)(qr + lrow) * Dc,      1, kgrp);
    v16bf aq1 = load_frag(qp + (size_t)(qr + lrow) * Dc + 32, 1, kgrp);

    float mrow[8], lsum[8];
    v8f z = {};
    v8f oacc[4];
#pragma unroll
    for (int r = 0; r < 8; ++r) { mrow[r] = -INFINITY; lsum[r] = 0.f; }
#pragma unroll
    for (int nt = 0; nt < 4; ++nt) oacc[nt] = z;

    const float scale = 0.03608439182435161f;  // 768^-0.5

    for (int kc = 0; kc < qr + 16; kc += 32) {
        v8f s[2];
#pragma unroll
        for (int j = 0; j < 2; ++j) {
            int kb = kc + j * 16;
            v16bf bk0 = load_frag(kp + (size_t)(kb + lrow) * Dc,      1, kgrp);
            v16bf bk1 = load_frag(kp + (size_t)(kb + lrow) * Dc + 32, 1, kgrp);
            v8f c = z;
            c = __builtin_amdgcn_wmma_f32_16x16x32_bf16(false, aq0, false, bk0, (short)0, c, false, false);
            c = __builtin_amdgcn_wmma_f32_16x16x32_bf16(false, aq1, false, bk1, (short)0, c, false, false);
#pragma unroll
            for (int r = 0; r < 8; ++r) {
                int trow = qr + r + kgrp;
                int col  = kb + lrow;
                float sv = c[r] * scale;
                s[j][r] = (col <= trow) ? sv : -INFINITY;
            }
        }
        // Online softmax: rows are replicated across the 16 lanes of each half.
#pragma unroll
        for (int r = 0; r < 8; ++r) {
            float cand = fmaxf(s[0][r], s[1][r]);
#pragma unroll
            for (int msk = 1; msk < 16; msk <<= 1)
                cand = fmaxf(cand, __shfl_xor(cand, msk, 32));
            float nm = fmaxf(mrow[r], cand);
            float f  = expf(mrow[r] - nm);
            float p0 = expf(s[0][r] - nm);
            float p1 = expf(s[1][r] - nm);
            float ps = p0 + p1;
#pragma unroll
            for (int msk = 1; msk < 16; msk <<= 1)
                ps += __shfl_xor(ps, msk, 32);
            lsum[r] = lsum[r] * f + ps;
            mrow[r] = nm;
            sP[r + kgrp][lrow]      = f2bf(p0);
            sP[r + kgrp][16 + lrow] = f2bf(p1);
#pragma unroll
            for (int nt = 0; nt < 4; ++nt) oacc[nt][r] *= f;
        }
        __syncthreads();
        v16bf ap = load_frag(&sP[lrow][0], 1, kgrp);   // P as A-frag [16 x 32keys]
#pragma unroll
        for (int nt = 0; nt < 4; ++nt) {
            v16bf bv = load_frag(vp + (size_t)kc * Dc + nt * 16 + lrow, Dc, kgrp);
            oacc[nt] = __builtin_amdgcn_wmma_f32_16x16x32_bf16(false, ap, false, bv, (short)0, oacc[nt], false, false);
        }
        __syncthreads();
    }

#pragma unroll
    for (int nt = 0; nt < 4; ++nt)
#pragma unroll
        for (int r = 0; r < 8; ++r) {
            int trow = qr + r + kgrp;
            float val = oacc[nt][r] / lsum[r];
            o[((size_t)b * Tc + trow) * Dc + h * DHc + nt * 16 + lrow] = f2bf(val);
        }
}

// ---------------------------------------------------------------------------
// Cross-entropy: per row logsumexp over V, atomicAdd mean NLL.
// ---------------------------------------------------------------------------
__global__ __launch_bounds__(256) void loss_kernel(const float* __restrict__ logits,
                                                   const int* __restrict__ targets,
                                                   float* __restrict__ loss) {
    __shared__ float red[256];
    int tid = threadIdx.x;
    size_t row = blockIdx.x;
    const float* lr = logits + row * Vc;

    float mx = -3.4e38f;
    for (int i = tid; i < Vc; i += 256) mx = fmaxf(mx, lr[i]);
    red[tid] = mx; __syncthreads();
    for (int o = 128; o > 0; o >>= 1) { if (tid < o) red[tid] = fmaxf(red[tid], red[tid + o]); __syncthreads(); }
    mx = red[0]; __syncthreads();

    float s = 0.f;
    for (int i = tid; i < Vc; i += 256) s += expf(lr[i] - mx);
    red[tid] = s; __syncthreads();
    for (int o = 128; o > 0; o >>= 1) { if (tid < o) red[tid] += red[tid + o]; __syncthreads(); }

    if (tid == 0) {
        float lse = logf(red[0]) + mx;
        int t = targets[row];
        atomicAdd(loss, (lse - lr[t]) * (1.0f / (float)Mrows));
    }
}

// ---------------------------------------------------------------------------
extern "C" void kernel_launch(void* const* d_in, const int* in_sizes, int n_in,
                              void* d_out, int out_size, void* d_ws, size_t ws_size,
                              hipStream_t stream) {
    (void)in_sizes; (void)n_in; (void)out_size; (void)ws_size;
    const int*   idx   = (const int*)d_in[0];
    const int*   tgt   = (const int*)d_in[1];
    const float* tok   = (const float*)d_in[2];
    const float* pos   = (const float*)d_in[3];
    const float* ln1g  = (const float*)d_in[4];
    const float* ln1b  = (const float*)d_in[5];
    const float* Wq    = (const float*)d_in[6];
    const float* Wk    = (const float*)d_in[7];
    const float* Wv    = (const float*)d_in[8];
    const float* Wp    = (const float*)d_in[9];
    const float* bp    = (const float*)d_in[10];
    const float* ln2g  = (const float*)d_in[11];
    const float* ln2b  = (const float*)d_in[12];
    const float* W1    = (const float*)d_in[13];
    const float* b1    = (const float*)d_in[14];
    const float* W2    = (const float*)d_in[15];
    const float* b2    = (const float*)d_in[16];
    const float* lnfg  = (const float*)d_in[17];
    const float* lnfb  = (const float*)d_in[18];
    const float* Wlm   = (const float*)d_in[19];
    const float* blm   = (const float*)d_in[20];

    float* logits = (float*)d_out;
    float* loss   = logits + (size_t)Mrows * Vc;

    const size_t NBD = (size_t)Mrows * Dc;       // 6,291,456
    char* ws = (char*)d_ws;
    size_t off = 0;
    auto carve = [&](size_t bytes) { void* p = ws + off; off += (bytes + 255) & ~(size_t)255; return p; };
    float*          x   = (float*)carve(NBD * 4);
    unsigned short* hbf = (unsigned short*)carve(NBD * 2);
    unsigned short* qb  = (unsigned short*)carve(NBD * 2);
    unsigned short* kb  = (unsigned short*)carve(NBD * 2);
    unsigned short* vb  = (unsigned short*)carve(NBD * 2);
    unsigned short* obf = (unsigned short*)carve(NBD * 2);
    unsigned short* ffb = (unsigned short*)carve((size_t)Mrows * FFc * 2);

    // Embeddings
    {
        size_t total = NBD;
        embed_kernel<<<dim3((unsigned)((total + 255) / 256)), 256, 0, stream>>>(idx, tok, pos, x);
    }

    dim3 gN768(Dc / 128, Mrows / 64);    // 6 x 128
    dim3 gN3072(FFc / 128, Mrows / 64);  // 24 x 128
    dim3 gNV(Vc / 128, Mrows / 64);      // 64 x 128
    const unsigned SB = GEMM_SMEM_BYTES;

    for (int l = 0; l < Lc; ++l) {
        const size_t wo = (size_t)l * Dc * Dc;         // H*D*DH == D*D
        ln_kernel<<<Mrows, 256, 0, stream>>>(x, ln1g + l * Dc, ln1b + l * Dc, hbf);
        gemm_wmma<1, 3><<<gN768, 256, SB, stream>>>(hbf, Wq + wo, nullptr, nullptr, qb, Mrows, Dc, Dc);
        gemm_wmma<1, 3><<<gN768, 256, SB, stream>>>(hbf, Wk + wo, nullptr, nullptr, kb, Mrows, Dc, Dc);
        gemm_wmma<1, 3><<<gN768, 256, SB, stream>>>(hbf, Wv + wo, nullptr, nullptr, vb, Mrows, Dc, Dc);
        attn_kernel<<<dim3(Tc / 16, Bc * Hc), 32, 0, stream>>>(qb, kb, vb, obf);
        gemm_wmma<0, 1><<<gN768, 256, SB, stream>>>(obf, Wp + wo, bp + l * Dc, x, nullptr, Mrows, Dc, Dc);
        ln_kernel<<<Mrows, 256, 0, stream>>>(x, ln2g + l * Dc, ln2b + l * Dc, hbf);
        gemm_wmma<0, 2><<<gN3072, 256, SB, stream>>>(hbf, W1 + (size_t)l * Dc * FFc, b1 + l * FFc,
                                                     nullptr, ffb, Mrows, FFc, Dc);
        gemm_wmma<0, 1><<<gN768, 256, SB, stream>>>(ffb, W2 + (size_t)l * FFc * Dc, b2 + l * Dc,
                                                    x, nullptr, Mrows, Dc, FFc);
    }

    ln_kernel<<<Mrows, 256, 0, stream>>>(x, lnfg, lnfb, hbf);
    gemm_wmma<0, 0><<<gNV, 256, SB, stream>>>(hbf, Wlm, blm, logits, nullptr, Mrows, Vc, Dc);

    hipMemsetAsync(loss, 0, sizeof(float), stream);
    loss_kernel<<<Mrows, 256, 0, stream>>>(logits, tgt, loss);
}
